// HWNN_63153199120588
// MI455X (gfx1250) — matrix operand order repaired and checked by reading, not compiled, not run on hardware
//
#include <hip/hip_runtime.h>
#include <stdint.h>

// ---------------------------------------------------------------------------
// HWNN hypergraph wavelet filter for MI455X (gfx1250).
//   out = poly(Theta) . diag(f) . poly_t(Theta^T) . features . W
// Theta is symmetric, so all 4 N x N applications are Theta @ x.
// Strategy: Theta -> bf16 once (128 MB, fits 192 MB L2), then 4 WMMA-bf16
// GEMM passes [8192x8192] x [8192x64] with fused par[i]-scaled accumulation.
// Workspace requirement: ~132 MB (ThetaBF 128MB + 2x1MB bf16 ping-pong + 2MB fp32 acc).
// ---------------------------------------------------------------------------

#define NN   8192
#define FDIM 64

typedef __attribute__((ext_vector_type(8)))  __bf16   v8bf;
typedef __attribute__((ext_vector_type(16))) __bf16   v16bf;
typedef __attribute__((ext_vector_type(8)))  float    v8f;
typedef __attribute__((ext_vector_type(4)))  uint16_t v4u16;

// fp32 -> bf16 round-to-nearest-even (bit trick; NaN edge ignored — inputs finite)
static __device__ __forceinline__ uint16_t f2bf(float f) {
    union { float f; uint32_t u; } c; c.f = f;
    uint32_t u = c.u;
    u = (u + 0x7FFFu + ((u >> 16) & 1u)) >> 16;
    return (uint16_t)u;
}

// --- Kernel 0: Theta fp32 -> bf16 (64M elems, 4/thread) ---------------------
__global__ void __launch_bounds__(256) cvt_theta_kernel(const float* __restrict__ Tf,
                                                        uint16_t* __restrict__ Tb) {
    size_t i = ((size_t)blockIdx.x * 256 + threadIdx.x) * 4;
    const float4 v = *(const float4*)(Tf + i);
    v4u16 o;
    o.x = f2bf(v.x); o.y = f2bf(v.y); o.z = f2bf(v.z); o.w = f2bf(v.w);
    *(v4u16*)(Tb + i) = o;
}

// --- Kernel 1: X = features @ weight; acc = par[3]*X; Xt(bf16, col-major) ---
__global__ void __launch_bounds__(256) xw_kernel(const float* __restrict__ features,
                                                 const float* __restrict__ weight,
                                                 const float* __restrict__ par,
                                                 float* __restrict__ accBuf,
                                                 uint16_t* __restrict__ Xt) {
    __shared__ float w[FDIM * FDIM];
    for (int i = threadIdx.x; i < FDIM * FDIM; i += 256) w[i] = weight[i];
    __syncthreads();
    const int row = blockIdx.x * 4 + (threadIdx.x >> 6);
    const int n   = threadIdx.x & 63;
    const float* f = features + (size_t)row * FDIM;
    float s = 0.f;
#pragma unroll
    for (int k = 0; k < FDIM; ++k) s += f[k] * w[k * FDIM + n];
    accBuf[(size_t)row * FDIM + n]   = par[3] * s;
    Xt[(size_t)n * NN + row]         = f2bf(s);
}

// --- Kernel 2: Y = diag * acc; out = par[0]*Y; Yt(bf16, col-major) ----------
__global__ void __launch_bounds__(256) y_kernel(const float* __restrict__ accBuf,
                                                const float* __restrict__ diag,
                                                const float* __restrict__ par,
                                                float* __restrict__ out,
                                                uint16_t* __restrict__ Yt) {
    const int idx = blockIdx.x * 256 + threadIdx.x;   // 0 .. N*F-1
    const int m = idx >> 6, n = idx & 63;
    const float y = diag[m] * accBuf[idx];
    out[idx]                 = par[0] * y;
    Yt[(size_t)n * NN + m]   = f2bf(y);
}

// --- Kernel 3: D = Theta(bf16) @ B; accOut += par[pidx]*D; Dt = bf16(D)^T ---
// Block: 128 threads = 4 waves. Wave (w&1) picks M half (16 rows of the
// block's 32), (w>>1) picks N half (32 of 64 cols) -> 2 WMMA chains per wave.
// A-operand per-lane addressing per ISA "16-bit A 16x32" layout:
//   lane<16: row=lane,    K = {k..k+7, k+16..k+23}
//   lane>=16: row=lane-16, K = {k+8..k+15, k+24..k+31}
// B-operand per ISA 16-bit 32x16 layout: lane = column, same K set as A per
// half-wave, packed 2/VGPR -> two 16-byte loads from the column-major buffer.
__global__ void __launch_bounds__(128) theta_gemm_kernel(const uint16_t* __restrict__ Abf,
                                                         const uint16_t* __restrict__ Bt,
                                                         const float* __restrict__ par,
                                                         int pidx,
                                                         float* __restrict__ accOut,
                                                         uint16_t* __restrict__ Dt) {
    const int lane    = threadIdx.x & 31;
    const int wave    = threadIdx.x >> 5;
    const int l15     = lane & 15;
    const int halfsel = lane >> 4;                 // 0 or 1
    const int m0      = blockIdx.x * 32 + (wave & 1) * 16;
    const int nbase   = (wave >> 1) * 32;

    const uint16_t* Arow = Abf + (size_t)(m0 + l15) * NN;
    const uint16_t* Bc0  = Bt + (size_t)(nbase +      l15) * NN;
    const uint16_t* Bc1  = Bt + (size_t)(nbase + 16 + l15) * NN;
    const int aoff = halfsel * 8;
    const int boff = halfsel * 16;

    v8f acc0 = {}, acc1 = {};

#pragma unroll 4
    for (int k = 0; k < NN; k += 32) {
        // A tile (16x32 bf16)
        v8bf alo = *(const v8bf*)(Arow + k + aoff);
        v8bf ahi = *(const v8bf*)(Arow + k + 16 + aoff);
        v16bf a  = __builtin_shufflevector(alo, ahi,
                     0, 1, 2, 3, 4, 5, 6, 7, 8, 9, 10, 11, 12, 13, 14, 15);
        // B tiles (32x16 bf16): K = {k+8h..+7, k+16+8h..+7}, two 16-byte loads each
        v8bf b0lo = *(const v8bf*)(Bc0 + k + aoff);
        v8bf b0hi = *(const v8bf*)(Bc0 + k + 16 + aoff);
        v16bf b0 = __builtin_shufflevector(b0lo, b0hi,
                     0, 1, 2, 3, 4, 5, 6, 7, 8, 9, 10, 11, 12, 13, 14, 15);
        v8bf b1lo = *(const v8bf*)(Bc1 + k + aoff);
        v8bf b1hi = *(const v8bf*)(Bc1 + k + 16 + aoff);
        v16bf b1 = __builtin_shufflevector(b1lo, b1hi,
                     0, 1, 2, 3, 4, 5, 6, 7, 8, 9, 10, 11, 12, 13, 14, 15);

        acc0 = __builtin_amdgcn_wmma_f32_16x16x32_bf16(false, a, false, b0,
                                                       (short)0, acc0, false, false);
        acc1 = __builtin_amdgcn_wmma_f32_16x16x32_bf16(false, a, false, b1,
                                                       (short)0, acc1, false, false);

        // stay 4 KB ahead in this A row; wrap keeps the address in-bounds
        __builtin_prefetch(Arow + ((k + 2048) & (NN - 1)), 0, 1);
    }

    const float coef = par[pidx];
    const int n0 = nbase + l15;
    const int n1 = nbase + 16 + l15;
#pragma unroll
    for (int r = 0; r < 8; ++r) {
        const int m = m0 + r + halfsel * 8;        // C/D layout: VGPR r, lane half
        accOut[(size_t)m * FDIM + n0] += coef * acc0[r];
        accOut[(size_t)m * FDIM + n1] += coef * acc1[r];
        Dt[(size_t)n0 * NN + m] = f2bf(acc0[r]);
        Dt[(size_t)n1 * NN + m] = f2bf(acc1[r]);
    }
}

// ---------------------------------------------------------------------------
extern "C" void kernel_launch(void* const* d_in, const int* in_sizes, int n_in,
                              void* d_out, int out_size, void* d_ws, size_t ws_size,
                              hipStream_t stream) {
    (void)in_sizes; (void)n_in; (void)out_size; (void)ws_size;
    const float* features = (const float*)d_in[0];   // [8192,64]
    const float* Theta    = (const float*)d_in[1];   // [8192,8192]
    const float* diag     = (const float*)d_in[2];   // [8192]
    const float* par      = (const float*)d_in[3];   // [6]
    const float* weight   = (const float*)d_in[4];   // [64,64]
    float* out = (float*)d_out;                      // [8192,64]

    char* ws = (char*)d_ws;
    uint16_t* thetaBF = (uint16_t*)(ws);                                   // 128 MB
    uint16_t* Bt0     = (uint16_t*)(ws + (size_t)NN * NN * 2);             // 1 MB
    uint16_t* Bt1     = (uint16_t*)(ws + (size_t)NN * NN * 2 + (size_t)NN * FDIM * 2);
    float*    accBuf  = (float*)   (ws + (size_t)NN * NN * 2 + (size_t)NN * FDIM * 4);

    const dim3 gemmGrid(NN / 32), gemmBlk(128);

    // Theta -> bf16 (resident in L2 for the remaining passes)
    cvt_theta_kernel<<<(NN * (size_t)NN) / (256 * 4), 256, 0, stream>>>(Theta, thetaBF);

    // X = features @ W ; acc = par[3]*X ; Xt bf16
    xw_kernel<<<NN / 4, 256, 0, stream>>>(features, weight, par, accBuf, Bt0);

    // acc += par[4]*(T X)  ;  acc += par[5]*(T^2 X)
    theta_gemm_kernel<<<gemmGrid, gemmBlk, 0, stream>>>(thetaBF, Bt0, par, 4, accBuf, Bt1);
    theta_gemm_kernel<<<gemmGrid, gemmBlk, 0, stream>>>(thetaBF, Bt1, par, 5, accBuf, Bt0);

    // Y = diag * acc ; out = par[0]*Y ; Yt bf16 (overwrites Bt0 fully)
    y_kernel<<<(NN * FDIM) / 256, 256, 0, stream>>>(accBuf, diag, par, out, Bt0);

    // out += par[1]*(T Y) ; out += par[2]*(T^2 Y)
    theta_gemm_kernel<<<gemmGrid, gemmBlk, 0, stream>>>(thetaBF, Bt0, par, 1, out, Bt1);
    theta_gemm_kernel<<<gemmGrid, gemmBlk, 0, stream>>>(thetaBF, Bt1, par, 2, out, Bt0);
}
